// MultiHeadAttention_61787399520262
// MI455X (gfx1250) — matrix operand order repaired and checked by reading, not compiled, not run on hardware
//
#include <hip/hip_runtime.h>

typedef __attribute__((ext_vector_type(16))) __bf16 v16bf;
typedef __attribute__((ext_vector_type(8)))  float  v8f;
typedef __attribute__((ext_vector_type(8)))  __bf16 bf8v;   // 16 bytes

#define NT   32768   // B*T
#define Cdim 384
#define Ddim 64
#define Hn   6
#define Tn   256
#define Bn   128
#define TSTR 80      // padded stride (elements) for transposed LDS tiles; 160B rows keep 16B align

// ---------- helpers ----------

__device__ __forceinline__ __bf16 f2bf(float f) {
  unsigned u = __builtin_bit_cast(unsigned, f);
  unsigned r = u + 0x7fffu + ((u >> 16) & 1u);   // round-to-nearest-even
  unsigned short hh = (unsigned short)(r >> 16);
  return __builtin_bit_cast(__bf16, hh);
}

// Async global->LDS 16B copy (GLOBAL_LOAD_ASYNC_TO_LDS_B128, ASYNCcnt-tracked).
// LDS destination = low 32 bits of the generic pointer.
__device__ __forceinline__ void async_copy_b128(void* lds_dst, const void* gsrc) {
  unsigned l = (unsigned)(unsigned long long)lds_dst;
  asm volatile("global_load_async_to_lds_b128 %0, %1, off"
               :: "v"(l), "v"(gsrc) : "memory");
}
__device__ __forceinline__ void wait_async() {
  asm volatile("s_wait_asynccnt 0x0" ::: "memory");
}

// A-matrix fragment (16x32 bf16, MxK) from row-major LDS A[m*lda + k].
// Lane map (ISA 7.12.2): elems 0..7 -> k = 8*half .. +7 ; elems 8..15 -> k = 16+8*half .. +7
__device__ __forceinline__ v16bf load_afrag(const __bf16* A, int lda, int lane) {
  const int m = lane & 15, half = lane >> 4;
  const bf8v lo = *(const bf8v*)&A[m * lda + 8 * half];
  const bf8v hi = *(const bf8v*)&A[m * lda + 16 + 8 * half];
  return __builtin_shufflevector(lo, hi, 0, 1, 2, 3, 4, 5, 6, 7, 8, 9, 10, 11, 12, 13, 14, 15);
}

// B-matrix fragment (32x16 bf16, KxN) from a K-contiguous (transposed) source:
// B[k][n] = S[(nbase+n)*stride + kbase + k]. Lane map: elems 0..7 -> k = 8*half .. +7 ; elems 8..15 -> k = 16+8*half .. +7
__device__ __forceinline__ v16bf load_bfrag_t(const __bf16* S, int stride, int lane, int nbase, int kbase) {
  const int n = lane & 15, half = lane >> 4;
  const __bf16* p = &S[(nbase + n) * stride + kbase + 8 * half];
  const bf8v lo = *(const bf8v*)p;
  const bf8v hi = *(const bf8v*)(p + 16);
  return __builtin_shufflevector(lo, hi, 0, 1, 2, 3, 4, 5, 6, 7, 8, 9, 10, 11, 12, 13, 14, 15);
}

__device__ __forceinline__ v8f wmma_bf16(v16bf a, v16bf b, v8f c) {
  return __builtin_amdgcn_wmma_f32_16x16x32_bf16(false, a, false, b, (short)0, c, false, false);
}

// reductions across the 16 lanes of a half-wave (xor masks stay inside each group of 16)
__device__ __forceinline__ float hmax16(float v) {
  v = fmaxf(v, __shfl_xor(v, 1, 32));
  v = fmaxf(v, __shfl_xor(v, 2, 32));
  v = fmaxf(v, __shfl_xor(v, 4, 32));
  v = fmaxf(v, __shfl_xor(v, 8, 32));
  return v;
}
__device__ __forceinline__ float hsum16(float v) {
  v += __shfl_xor(v, 1, 32);
  v += __shfl_xor(v, 2, 32);
  v += __shfl_xor(v, 4, 32);
  v += __shfl_xor(v, 8, 32);
  return v;
}

// ---------- kernel 0: f32 -> bf16 ----------
__global__ __launch_bounds__(256) void cvt_bf16_kernel(const float* __restrict__ in,
                                                       __bf16* __restrict__ out, int nel) {
  int i = blockIdx.x * 256 + threadIdx.x;
  if (i < nel) out[i] = f2bf(in[i]);
}

// ---------- kernel 1: QKV projection GEMM ----------
// per block: 64 rows (tokens) x 64 cols (= D) for one head, one of {q,k,v}; K-tile = 64
__global__ __launch_bounds__(128) void qkv_gemm_kernel(
    const __bf16* __restrict__ xh,
    const __bf16* __restrict__ wq, const __bf16* __restrict__ wk, const __bf16* __restrict__ wv,
    __bf16* __restrict__ qb, __bf16* __restrict__ kb, __bf16* __restrict__ vb) {
  const int m0 = blockIdx.x * 64;
  const int h  = blockIdx.y;
  const int which = blockIdx.z;
  const __bf16* __restrict__ w = (which == 0) ? wq : (which == 1) ? wk : wv;
  __bf16* __restrict__ outp    = (which == 0) ? qb : (which == 1) ? kb : vb;

  __shared__ alignas(16) __bf16 As[64 * 64];        // row-major [m][k]
  __shared__ alignas(16) __bf16 Bt[64 * TSTR];      // transposed [n][k]

  const int tid = threadIdx.x, lane = tid & 31, wid = tid >> 5;
  v8f acc[4] = {};

  for (int k0 = 0; k0 < Cdim; k0 += 64) {
    __syncthreads();
    // A tile: async global->LDS (no VGPR round trip)
#pragma unroll
    for (int it = 0; it < 4; ++it) {
      const int v = tid + it * 128;
      const int r = v >> 3, cv = (v & 7) * 8;
      async_copy_b128(&As[r * 64 + cv], &xh[(size_t)(m0 + r) * Cdim + k0 + cv]);
    }
    // B tile: bulk-load into registers, then transpose-store to LDS
    bf8v bw[4];
#pragma unroll
    for (int it = 0; it < 4; ++it) {
      const int v = tid + it * 128;
      const int r = v >> 3, cv = (v & 7) * 8;
      bw[it] = *(const bf8v*)&w[((size_t)h * Cdim + k0 + r) * Ddim + cv];
    }
#pragma unroll
    for (int it = 0; it < 4; ++it) {
      const int v = tid + it * 128;
      const int r = v >> 3, cv = (v & 7) * 8;
#pragma unroll
      for (int i = 0; i < 8; ++i) Bt[(cv + i) * TSTR + r] = bw[it][i];
    }
    if (k0 + 64 < Cdim)  // warm cache for next A panel -> global_prefetch_b8
      __builtin_prefetch(&xh[(size_t)(m0 + (tid >> 1)) * Cdim + k0 + 64], 0, 1);
    wait_async();
    __syncthreads();
    const v16bf a0 = load_afrag(&As[wid * 16 * 64], 64, lane);
    const v16bf a1 = load_afrag(&As[wid * 16 * 64 + 32], 64, lane);
#pragma unroll
    for (int nb = 0; nb < 4; ++nb) {
      acc[nb] = wmma_bf16(a0, load_bfrag_t(Bt, TSTR, lane, nb * 16, 0),  acc[nb]);
      acc[nb] = wmma_bf16(a1, load_bfrag_t(Bt, TSTR, lane, nb * 16, 32), acc[nb]);
    }
  }

  const float scale = (which == 0) ? 0.125f : 1.0f;   // fold 1/sqrt(D) into Q
  const int n = lane & 15, half = lane >> 4;
#pragma unroll
  for (int nb = 0; nb < 4; ++nb)
#pragma unroll
    for (int j = 0; j < 8; ++j) {
      const int row = m0 + wid * 16 + j + 8 * half;   // = b*T + t
      const int col = nb * 16 + n;                    // d
      outp[((size_t)h * NT + row) * Ddim + col] = f2bf(acc[nb][j] * scale);
    }
}

// ---------- kernel 2: causal flash attention ----------
// grid: (H*B, T/64); block = 128 threads (4 waves), each wave owns 16 query rows
__global__ __launch_bounds__(128) void attn_kernel(
    const __bf16* __restrict__ qb, const __bf16* __restrict__ kb,
    const __bf16* __restrict__ vb, __bf16* __restrict__ ao) {
  const int hb = blockIdx.x;       // h*Bn + b
  const int h  = hb >> 7;
  const int b  = hb & 127;
  const int qblk  = blockIdx.y;
  const int qbase = qblk * 64;
  const size_t base = (size_t)hb * Tn * Ddim;

  __shared__ alignas(16) __bf16 Qs[64 * 64];        // [q][d]  row-major
  __shared__ alignas(16) __bf16 Ks[64 * 64];        // [key][d] row-major (K-contig for S=Q*K^T)
  __shared__ alignas(16) __bf16 Vt[64 * TSTR];      // [d][key] transposed (K-contig for O=P*V)
  __shared__ alignas(16) __bf16 Ps[4][16 * 64];     // per-wave P tile, row-major [m][key]

  const int tid = threadIdx.x, lane = tid & 31, wid = tid >> 5;
  const int n = lane & 15, half = lane >> 4;

#pragma unroll
  for (int it = 0; it < 4; ++it) {                   // Q tile: async global->LDS
    const int v = tid + it * 128;
    const int r = v >> 3, cv = (v & 7) * 8;
    async_copy_b128(&Qs[r * 64 + cv], &qb[base + (size_t)(qbase + r) * Ddim + cv]);
  }
  wait_async();
  __syncthreads();
  const v16bf qa0 = load_afrag(&Qs[wid * 16 * 64], 64, lane);        // d 0..31
  const v16bf qa1 = load_afrag(&Qs[wid * 16 * 64 + 32], 64, lane);   // d 32..63

  float rmax[8], rsum[8];
#pragma unroll
  for (int j = 0; j < 8; ++j) { rmax[j] = -__builtin_inff(); rsum[j] = 0.f; }
  v8f oacc[4] = {};

  for (int kt = 0; kt <= qblk; ++kt) {       // uniform bound -> barriers legal
    const int kbase = kt * 64;
    __syncthreads();
    // K tile: async global->LDS ; V tile: bulk-load then transpose-store
#pragma unroll
    for (int it = 0; it < 4; ++it) {
      const int v = tid + it * 128;
      const int r = v >> 3, cv = (v & 7) * 8;
      async_copy_b128(&Ks[r * 64 + cv], &kb[base + (size_t)(kbase + r) * Ddim + cv]);
    }
    bf8v vv[4];
#pragma unroll
    for (int it = 0; it < 4; ++it) {
      const int v = tid + it * 128;
      const int r = v >> 3, cv = (v & 7) * 8;
      vv[it] = *(const bf8v*)&vb[base + (size_t)(kbase + r) * Ddim + cv];
    }
#pragma unroll
    for (int it = 0; it < 4; ++it) {
      const int v = tid + it * 128;
      const int r = v >> 3, cv = (v & 7) * 8;
#pragma unroll
      for (int i = 0; i < 8; ++i) Vt[(cv + i) * TSTR + r] = vv[it][i];
    }
    wait_async();
    __syncthreads();

    // S = Q * K^T  (16 queries x 64 keys per wave); B[k=d][n=key] = Ks[key*64 + d] (K-contig)
    v8f s[4];
#pragma unroll
    for (int nb = 0; nb < 4; ++nb) {
      v8f c = {};
      c = wmma_bf16(qa0, load_bfrag_t(Ks, 64, lane, nb * 16, 0), c);
      c = wmma_bf16(qa1, load_bfrag_t(Ks, 64, lane, nb * 16, 32), c);
      s[nb] = c;
    }

    if (kt == qblk) {   // diagonal tile: per-element causal mask
#pragma unroll
      for (int nb = 0; nb < 4; ++nb)
#pragma unroll
        for (int j = 0; j < 8; ++j) {
          const int key  = kbase + nb * 16 + n;
          const int qrow = qbase + wid * 16 + j + 8 * half;
          if (key > qrow) s[nb][j] = -__builtin_inff();
        }
    }

    // online softmax (per-row stats; rows span the 16 lanes of a half-wave)
#pragma unroll
    for (int j = 0; j < 8; ++j) {
      float t = fmaxf(fmaxf(s[0][j], s[1][j]), fmaxf(s[2][j], s[3][j]));
      t = hmax16(t);
      const float nm   = fmaxf(rmax[j], t);
      const float corr = __expf(rmax[j] - nm);
      float psum = 0.f;
#pragma unroll
      for (int nb = 0; nb < 4; ++nb) {
        const float p = __expf(s[nb][j] - nm);
        psum += p;
        Ps[wid][(j + 8 * half) * 64 + nb * 16 + n] = f2bf(p);
      }
      psum = hsum16(psum);
      rsum[j] = rsum[j] * corr + psum;
      rmax[j] = nm;
#pragma unroll
      for (int db = 0; db < 4; ++db) oacc[db][j] *= corr;
    }
    __syncthreads();   // make Ps/Vt visible, keep waves in lockstep

    // O += P * V ; B[k=key][n=d] = Vt[d*TSTR + key] (K-contig)
#pragma unroll
    for (int db = 0; db < 4; ++db) {
#pragma unroll
      for (int kc = 0; kc < 2; ++kc) {
        const v16bf pa = load_afrag(&Ps[wid][kc * 32], 64, lane);
        const v16bf bv = load_bfrag_t(Vt, TSTR, lane, db * 16, kc * 32);
        oacc[db] = wmma_bf16(pa, bv, oacc[db]);
      }
    }
  }

  // normalize and write concat-head layout [b*T+t][h*D+d] (bf16, feeds out-proj GEMM)
#pragma unroll
  for (int db = 0; db < 4; ++db)
#pragma unroll
    for (int j = 0; j < 8; ++j) {
      const int bt  = b * Tn + qbase + wid * 16 + j + 8 * half;
      const int col = h * Ddim + db * 16 + n;
      ao[(size_t)bt * Cdim + col] = f2bf(oacc[db][j] / rsum[j]);
    }
}

// ---------- kernel 3: output projection + bias ----------
__global__ __launch_bounds__(128) void proj_kernel(
    const __bf16* __restrict__ ah, const __bf16* __restrict__ wph,
    const float* __restrict__ bp, float* __restrict__ out) {
  const int m0 = blockIdx.x * 64;
  const int n0 = blockIdx.y * 64;
  __shared__ alignas(16) __bf16 As[64 * 64];
  __shared__ alignas(16) __bf16 Bt[64 * TSTR];
  const int tid = threadIdx.x, lane = tid & 31, wid = tid >> 5;
  v8f acc[4] = {};
  for (int k0 = 0; k0 < Cdim; k0 += 64) {
    __syncthreads();
#pragma unroll
    for (int it = 0; it < 4; ++it) {
      const int v = tid + it * 128;
      const int r = v >> 3, cv = (v & 7) * 8;
      async_copy_b128(&As[r * 64 + cv], &ah[(size_t)(m0 + r) * Cdim + k0 + cv]);
    }
    bf8v bw[4];
#pragma unroll
    for (int it = 0; it < 4; ++it) {
      const int v = tid + it * 128;
      const int r = v >> 3, cv = (v & 7) * 8;
      bw[it] = *(const bf8v*)&wph[(size_t)(k0 + r) * Cdim + n0 + cv];
    }
#pragma unroll
    for (int it = 0; it < 4; ++it) {
      const int v = tid + it * 128;
      const int r = v >> 3, cv = (v & 7) * 8;
#pragma unroll
      for (int i = 0; i < 8; ++i) Bt[(cv + i) * TSTR + r] = bw[it][i];
    }
    if (k0 + 64 < Cdim)
      __builtin_prefetch(&ah[(size_t)(m0 + (tid >> 1)) * Cdim + k0 + 64], 0, 1);
    wait_async();
    __syncthreads();
    const v16bf a0 = load_afrag(&As[wid * 16 * 64], 64, lane);
    const v16bf a1 = load_afrag(&As[wid * 16 * 64 + 32], 64, lane);
#pragma unroll
    for (int nb = 0; nb < 4; ++nb) {
      acc[nb] = wmma_bf16(a0, load_bfrag_t(Bt, TSTR, lane, nb * 16, 0),  acc[nb]);
      acc[nb] = wmma_bf16(a1, load_bfrag_t(Bt, TSTR, lane, nb * 16, 32), acc[nb]);
    }
  }
  const int n = lane & 15, half = lane >> 4;
#pragma unroll
  for (int nb = 0; nb < 4; ++nb)
#pragma unroll
    for (int j = 0; j < 8; ++j) {
      const int row = m0 + wid * 16 + j + 8 * half;
      const int col = n0 + nb * 16 + n;
      out[(size_t)row * Cdim + col] = acc[nb][j] + bp[col];
    }
}

// ---------- host launch ----------
extern "C" void kernel_launch(void* const* d_in, const int* in_sizes, int n_in,
                              void* d_out, int out_size, void* d_ws, size_t ws_size,
                              hipStream_t stream) {
  const float* x  = (const float*)d_in[0];
  const float* Wq = (const float*)d_in[1];
  const float* Wk = (const float*)d_in[2];
  const float* Wv = (const float*)d_in[3];
  const float* Wp = (const float*)d_in[4];
  const float* bp = (const float*)d_in[5];

  char* ws = (char*)d_ws;
  size_t off = 0;
  auto carve = [&](size_t elems) -> __bf16* {
    void* p = ws + off;
    off += (elems * 2 + 255) & ~(size_t)255;
    return (__bf16*)p;
  };

  const size_t nx  = (size_t)NT * Cdim;          // 12.58M
  const size_t nw  = (size_t)Hn * Cdim * Ddim;   // 147456
  const size_t np  = (size_t)Cdim * Cdim;        // 147456
  const size_t nhd = (size_t)Hn * NT * Ddim;     // 12.58M

  __bf16* xh  = carve(nx);
  __bf16* wqh = carve(nw);
  __bf16* wkh = carve(nw);
  __bf16* wvh = carve(nw);
  __bf16* wph = carve(np);
  __bf16* qb  = carve(nhd);
  __bf16* kb  = carve(nhd);
  __bf16* vb  = carve(nhd);
  __bf16* ah  = xh;   // reuse: xh is dead after QKV GEMM (stream-ordered)

  cvt_bf16_kernel<<<(int)((nx + 255) / 256), 256, 0, stream>>>(x, xh, (int)nx);
  cvt_bf16_kernel<<<(int)((nw + 255) / 256), 256, 0, stream>>>(Wq, wqh, (int)nw);
  cvt_bf16_kernel<<<(int)((nw + 255) / 256), 256, 0, stream>>>(Wk, wkh, (int)nw);
  cvt_bf16_kernel<<<(int)((nw + 255) / 256), 256, 0, stream>>>(Wv, wvh, (int)nw);
  cvt_bf16_kernel<<<(int)((np + 255) / 256), 256, 0, stream>>>(Wp, wph, (int)np);

  qkv_gemm_kernel<<<dim3(NT / 64, Hn, 3), 128, 0, stream>>>(xh, wqh, wkh, wvh, qb, kb, vb);
  attn_kernel<<<dim3(Hn * Bn, Tn / 64), 128, 0, stream>>>(qb, kb, vb, ah);
  proj_kernel<<<dim3(NT / 64, Cdim / 64), 128, 0, stream>>>(ah, wph, bp, (float*)d_out);
}
